// Proxy_Anchor_51427938402969
// MI455X (gfx1250) — matrix operand order repaired and checked by reading, not compiled, not run on hardware
//
#include <hip/hip_runtime.h>
#include <hip/hip_bf16.h>
#include <math.h>

#define B_SZ   4096
#define EMB    512
#define FC     512
#define C_SZ   16384
#define ALPHA  32.0f
#define MRG    0.1f
#define THREADS 256
#define TILE_CLS 128   // classes (or fc cols) per block: 8 waves * 16

typedef __attribute__((ext_vector_type(16))) _Float16 v16h;
typedef __attribute__((ext_vector_type(8)))  _Float16 v8h;
typedef __attribute__((ext_vector_type(4)))  _Float16 v4h;
typedef __attribute__((ext_vector_type(8)))  float    v8f;

// Build the 16-bit A fragment (16x32, M=lane&15), contiguous K order:
// lanes 0-15: halves[0..15]=K[0..15]
// lanes16-31: halves[0..15]=K[16..31]
__device__ __forceinline__ v16h make_a_frag(const _Float16* row, int kk, int h) {
  v8h lo = *(const v8h*)(row + kk + 16 * h);
  v8h hi = *(const v8h*)(row + kk + 16 * h + 8);
  return __builtin_shufflevector(lo, hi, 0,1,2,3,4,5,6,7,8,9,10,11,12,13,14,15);
}

// ---------------- proxies: l2_norm -> f16 ----------------
__global__ void norm_proxies(const float* __restrict__ P, _Float16* __restrict__ Pn) {
  __shared__ float red[THREADS];
  int c = blockIdx.x, t = threadIdx.x;
  const float* row = P + (size_t)c * EMB;
  float a = row[t], b = row[t + 256];
  red[t] = a * a + b * b;
  __syncthreads();
  for (int o = 128; o > 0; o >>= 1) { if (t < o) red[t] += red[t + o]; __syncthreads(); }
  float inv = 1.0f / sqrtf(red[0] + 1e-12f);
  Pn[(size_t)c * EMB + t]       = (_Float16)(a * inv);
  Pn[(size_t)c * EMB + t + 256] = (_Float16)(b * inv);
}

// ---------------- fc: Y = X @ Wt + b  (WMMA f16, f32 out) ----------------
__global__ void fc_wmma(const float* __restrict__ X, const float* __restrict__ W,
                        const float* __restrict__ bias, float* __restrict__ Y) {
  extern __shared__ char smem[];
  _Float16* sW = (_Float16*)smem;                                // 128*512 f16
  _Float16* sX = (_Float16*)(smem + (size_t)TILE_CLS * EMB * 2); // 16*512 f16
  int tid = threadIdx.x;
  int j0 = blockIdx.x * TILE_CLS;
  int mBase = blockIdx.y * (B_SZ / gridDim.y);
  int nTiles = (B_SZ / gridDim.y) / 16;

  // stage W tile, f32 -> f16
  const float4* wsrc = (const float4*)(W + (size_t)j0 * EMB);
  v4h* wdst = (v4h*)sW;
#pragma unroll
  for (int j = 0; j < (TILE_CLS * EMB / 4) / THREADS; ++j) {
    int idx = tid + j * THREADS;
    float4 f = wsrc[idx];
    v4h hh; hh[0] = (_Float16)f.x; hh[1] = (_Float16)f.y; hh[2] = (_Float16)f.z; hh[3] = (_Float16)f.w;
    wdst[idx] = hh;
  }

  int w = tid >> 5, lane = tid & 31;
  int col = lane & 15, h = lane >> 4;
  int jcol = j0 + w * 16 + col;
  float bj = bias[jcol];

  for (int mt = 0; mt < nTiles; ++mt) {
    int m0 = mBase + mt * 16;
    __syncthreads();
    const float4* xsrc = (const float4*)(X + (size_t)m0 * EMB);
    v4h* xdst = (v4h*)sX;
#pragma unroll
    for (int j = 0; j < (16 * EMB / 4) / THREADS; ++j) {
      int idx = tid + j * THREADS;
      float4 f = xsrc[idx];
      v4h hh; hh[0] = (_Float16)f.x; hh[1] = (_Float16)f.y; hh[2] = (_Float16)f.z; hh[3] = (_Float16)f.w;
      xdst[idx] = hh;
    }
    __syncthreads();

    v8f acc = {};
#pragma unroll
    for (int kb = 0; kb < EMB / 32; ++kb) {
      int kk = kb * 32;
      v16h af = make_a_frag(sX + (size_t)col * EMB, kk, h);
      v16h bf = *(const v16h*)(sW + (size_t)(w * 16 + col) * EMB + kk + 16 * h);
      acc = __builtin_amdgcn_wmma_f32_16x16x32_f16(false, af, false, bf, (short)0, acc, false, false);
    }
#pragma unroll
    for (int v = 0; v < 8; ++v) {
      int m = m0 + v + 8 * h;
      Y[(size_t)m * FC + jcol] = acc[v] + bj;
    }
  }
}

// ---------------- normalize -> relu -> l2_norm -> f16 ----------------
__global__ void norm_x(const float* __restrict__ Y, _Float16* __restrict__ Xn) {
  __shared__ float red[THREADS];
  int b = blockIdx.x, t = threadIdx.x;
  const float* row = Y + (size_t)b * FC;
  float a = row[t], c = row[t + 256];
  red[t] = a * a + c * c;
  __syncthreads();
  for (int o = 128; o > 0; o >>= 1) { if (t < o) red[t] += red[t + o]; __syncthreads(); }
  float inv = 1.0f / fmaxf(sqrtf(red[0]), 1e-12f);   // F.normalize semantics
  __syncthreads();
  float r0 = fmaxf(a * inv, 0.0f), r1 = fmaxf(c * inv, 0.0f);
  red[t] = r0 * r0 + r1 * r1;
  __syncthreads();
  for (int o = 128; o > 0; o >>= 1) { if (t < o) red[t] += red[t + o]; __syncthreads(); }
  float inv2 = 1.0f / sqrtf(red[0] + 1e-12f);        // l2_norm semantics
  Xn[(size_t)b * FC + t]       = (_Float16)(r0 * inv2);
  Xn[(size_t)b * FC + t + 256] = (_Float16)(r1 * inv2);
}

// ------- fused cos-GEMM + exp + per-class reduction (B in registers) -------
__global__ void pa_main(const _Float16* __restrict__ Xn, const _Float16* __restrict__ Pn,
                        const int* __restrict__ T,
                        float* __restrict__ Psum, float* __restrict__ Nsum) {
  extern __shared__ char smem[];
  _Float16* sX = (_Float16*)smem;               // 16*512 f16 = 16 KB
  int*      sT = (int*)(smem + 16 * EMB * 2);   // 16 ints
  int tid = threadIdx.x;
  int c0 = blockIdx.x * TILE_CLS;
  int rowsPerBlock = B_SZ / gridDim.y;
  int mBase = blockIdx.y * rowsPerBlock;
  int nTiles = rowsPerBlock / 16;

  int w = tid >> 5, lane = tid & 31;
  int col = lane & 15, h = lane >> 4;
  int cls = c0 + w * 16 + col;

  // Register-cache ALL B fragments for this wave's 16 classes (full K=512):
  // frag[kb] element j = Pn[cls][kb*32 + 16*h + j]  (B column=lane&15, K split by lane half)
  v16h bfrag[EMB / 32];
  {
    const _Float16* prow = Pn + (size_t)cls * EMB + 16 * h;
#pragma unroll
    for (int kb = 0; kb < EMB / 32; ++kb)
      bfrag[kb] = *(const v16h*)(prow + kb * 32);
  }

  float posAcc = 0.0f, negAcc = 0.0f;

  // stage tile 0
  {
    const uint4* xsrc = (const uint4*)(Xn + (size_t)mBase * EMB);
    uint4* xdst = (uint4*)sX;
#pragma unroll
    for (int j = 0; j < (16 * EMB / 8) / THREADS; ++j)
      xdst[tid + j * THREADS] = xsrc[tid + j * THREADS];
    if (tid < 16) sT[tid] = T[mBase + tid];
  }
  __syncthreads();

  for (int mt = 0; mt < nTiles; ++mt) {
    // issue next tile's global loads into registers before the WMMA chain
    uint4 nx[4];
    int nT = 0;
    bool hasNext = (mt + 1 < nTiles);
    if (hasNext) {
      int m1 = mBase + (mt + 1) * 16;
      const uint4* xsrc = (const uint4*)(Xn + (size_t)m1 * EMB);
#pragma unroll
      for (int j = 0; j < 4; ++j) nx[j] = xsrc[tid + j * THREADS];
      if (tid < 16) nT = T[m1 + tid];
    }

    v8f acc = {};
#pragma unroll
    for (int kb = 0; kb < EMB / 32; ++kb) {
      v16h af = make_a_frag(sX + (size_t)col * EMB, kb * 32, h);
      acc = __builtin_amdgcn_wmma_f32_16x16x32_f16(false, af, false, bfrag[kb],
                                                   (short)0, acc, false, false);
    }
    // D element (M = v + 8*h, N = col); fuse exp + masked accumulate
#pragma unroll
    for (int v = 0; v < 8; ++v) {
      float cosv = acc[v];
      int m = v + 8 * h;
      if (sT[m] == cls) posAcc += __expf(-ALPHA * (cosv - MRG));
      else              negAcc += __expf( ALPHA * (cosv + MRG));
    }
    __syncthreads();
    if (hasNext) {
      uint4* xdst = (uint4*)sX;
#pragma unroll
      for (int j = 0; j < 4; ++j) xdst[tid + j * THREADS] = nx[j];
      if (tid < 16) sT[tid] = nT;
      __syncthreads();
    }
  }

  // combine the two half-wave copies of each class
  posAcc += __shfl_xor(posAcc, 16);
  negAcc += __shfl_xor(negAcc, 16);
  if (lane < 16) {
    atomicAdd(&Psum[c0 + w * 16 + lane], posAcc);
    atomicAdd(&Nsum[c0 + w * 16 + lane], negAcc);
  }
}

// ---------------- log1p sums, valid-proxy count, scalar output ----------------
__global__ void pa_finalize(const float* __restrict__ Psum, const float* __restrict__ Nsum,
                            float* __restrict__ out) {
  __shared__ float sP[THREADS], sN[THREADS], sV[THREADS];
  int t = threadIdx.x;
  float lp = 0.0f, ln = 0.0f, lv = 0.0f;
  for (int c = t; c < C_SZ; c += THREADS) {
    float p = Psum[c], n = Nsum[c];
    lp += log1pf(p);
    ln += log1pf(n);
    lv += (p > 0.0f) ? 1.0f : 0.0f;   // class has >=1 sample iff pos sum > 0
  }
  sP[t] = lp; sN[t] = ln; sV[t] = lv;
  __syncthreads();
  for (int o = 128; o > 0; o >>= 1) {
    if (t < o) { sP[t] += sP[t + o]; sN[t] += sN[t + o]; sV[t] += sV[t + o]; }
    __syncthreads();
  }
  if (t == 0) out[0] = sP[0] / fmaxf(sV[0], 1.0f) + sN[0] / (float)C_SZ;
}

extern "C" void kernel_launch(void* const* d_in, const int* in_sizes, int n_in,
                              void* d_out, int out_size, void* d_ws, size_t ws_size,
                              hipStream_t stream) {
  const float* X       = (const float*)d_in[0];
  const int*   T       = (const int*)  d_in[1];
  const float* proxies = (const float*)d_in[2];
  const float* fc_w    = (const float*)d_in[3];
  const float* fc_b    = (const float*)d_in[4];

  char* ws = (char*)d_ws;
  size_t off = 0;
  _Float16* Pn = (_Float16*)(ws + off); off += (size_t)C_SZ * EMB * 2;  // 16 MB
  _Float16* Xn = (_Float16*)(ws + off); off += (size_t)B_SZ * EMB * 2;  //  4 MB
  float*    Yf = (float*)   (ws + off); off += (size_t)B_SZ * FC * 4;   //  8 MB
  float*    Ps = (float*)   (ws + off); off += (size_t)C_SZ * 4;
  float*    Ns = (float*)   (ws + off); off += (size_t)C_SZ * 4;

  // zero the per-class accumulators each call (Ps and Ns are contiguous)
  hipMemsetAsync(Ps, 0, (size_t)2 * C_SZ * 4, stream);

  norm_proxies<<<C_SZ, THREADS, 0, stream>>>(proxies, Pn);

  size_t smemFc = (size_t)(TILE_CLS + 16) * EMB * 2;                    // 144 KB
  fc_wmma<<<dim3(FC / TILE_CLS, 32), THREADS, smemFc, stream>>>(X, fc_w, fc_b, Yf);

  norm_x<<<B_SZ, THREADS, 0, stream>>>(Yf, Xn);

  size_t smemMain = (size_t)16 * EMB * 2 + 16 * 4;                      // 16 KB + T tile
  pa_main<<<dim3(C_SZ / TILE_CLS, 8), THREADS, smemMain, stream>>>(Xn, Pn, T, Ps, Ns);

  pa_finalize<<<1, THREADS, 0, stream>>>(Ps, Ns, (float*)d_out);
}
